// LSTM_17635135717583
// MI455X (gfx1250) — matrix operand order repaired and checked by reading, not compiled, not run on hardware
//
#include <hip/hip_runtime.h>

#define B_SZ   64
#define T_SZ   512
#define I_SZ   1024
#define H_SZ   1024
#define N_KT   64          // K chunks of 32 (K_TOT = 2048)
#define NBLK   64          // each block owns 16 h-columns (all 4 gates)
#define NTHR   512         // 16 wave32s

typedef __attribute__((ext_vector_type(16))) _Float16 v16h;
typedef __attribute__((ext_vector_type(8)))  _Float16 v8h;
typedef __attribute__((ext_vector_type(8)))  float    v8f;
typedef __attribute__((ext_vector_type(4)))  float    f4;

__device__ __forceinline__ v16h mk16(v8h lo, v8h hi) {
    v16h r;
#pragma unroll
    for (int i = 0; i < 8; ++i) { r[i] = lo[i]; r[i + 8] = hi[i]; }
    return r;
}

// ---------------------------------------------------------------------------
// Pack fused [Wx;Wh] (2048 x 4096 f32, gate order i,f,g,o) into f16
// B-fragment-major layout: [ntg(256)][kt(64)][lane(32)][16 halves], where
// lane holds column n = lane&15, k = kt*32 + (lane>=16?16:0) + j  (j=0..15).
// ---------------------------------------------------------------------------
__global__ void pack_w(const float* __restrict__ Wx0, const float* __restrict__ Wx1,
                       const float* __restrict__ Wx2, const float* __restrict__ Wx3,
                       const float* __restrict__ Wh0, const float* __restrict__ Wh1,
                       const float* __restrict__ Wh2, const float* __restrict__ Wh3,
                       _Float16* __restrict__ Wp) {
    unsigned u    = blockIdx.x * blockDim.x + threadIdx.x;  // [0, 256*64*32)
    unsigned ntg  = u >> 11;
    unsigned kt   = (u >> 5) & 63u;
    unsigned lane = u & 31u;
    unsigned gate = ntg >> 6;
    unsigned col  = ((ntg & 63u) << 4) | (lane & 15u);
    unsigned kb   = kt * 32u + ((lane >> 4) << 4);
    const float* Wx = (gate == 0) ? Wx0 : (gate == 1) ? Wx1 : (gate == 2) ? Wx2 : Wx3;
    const float* Wh = (gate == 0) ? Wh0 : (gate == 1) ? Wh1 : (gate == 2) ? Wh2 : Wh3;
    _Float16* dst = Wp + (size_t)u * 16;
#pragma unroll
    for (int j = 0; j < 16; ++j) {
        unsigned k = kb + j;
        float v = (k < I_SZ) ? Wx[(size_t)k * H_SZ + col]
                             : Wh[(size_t)(k - I_SZ) * H_SZ + col];
        dst[j] = (_Float16)v;
    }
}

// Fuse biases, zero h0 (f16), reset barrier. Runs every launch (deterministic).
__global__ void init_misc(const float* __restrict__ bi0, const float* __restrict__ bi1,
                          const float* __restrict__ bi2, const float* __restrict__ bi3,
                          const float* __restrict__ bh0, const float* __restrict__ bh1,
                          const float* __restrict__ bh2, const float* __restrict__ bh3,
                          float* __restrict__ bias, _Float16* __restrict__ hbuf0,
                          unsigned* __restrict__ bar) {
    unsigned i = blockIdx.x * blockDim.x + threadIdx.x;  // 65536 threads
    hbuf0[i] = (_Float16)0.0f;
    if (i < 4096) {
        unsigned g = i >> 10, c = i & 1023u;
        const float* bi = (g == 0) ? bi0 : (g == 1) ? bi1 : (g == 2) ? bi2 : bi3;
        const float* bh = (g == 0) ? bh0 : (g == 1) ? bh1 : (g == 2) ? bh2 : bh3;
        bias[i] = bi[c] + bh[c];
    }
    if (i == 0) { bar[0] = 0u; bar[1] = 0u; }
}

// ---------------------------------------------------------------------------
// Persistent LSTM: 64 blocks x 512 threads, one grid barrier per timestep.
// All B fragments for the block (256 KB) cached in LDS for the whole kernel.
// K staged in 64-wide chunks (two WMMAs per block barrier).
// c state lives in registers; h passed between blocks in f16 via global ws.
// ---------------------------------------------------------------------------
__launch_bounds__(NTHR, 1)
__global__ void lstm_persistent(const float* __restrict__ x,
                                const _Float16* __restrict__ Wp,
                                const float* __restrict__ bias,
                                _Float16* __restrict__ hbuf,  // [2][64*1024] f16
                                float* __restrict__ out,      // h_T then c_T (f32)
                                unsigned* __restrict__ bar) {
    __shared__ __align__(16) _Float16 Bl[4 * 64 * 32 * 16];  // 256 KB B cache
    __shared__ __align__(16) _Float16 As[2][64 * 72];        // 18 KB A chunk (stride 72)
    __shared__ __align__(16) float    G[4 * 64 * 17];        // 17 KB gate exchange

    const int tid  = threadIdx.x;
    const int lane = tid & 31;
    const int wv   = tid >> 5;          // 0..15
    const int gate = wv >> 2;
    const int mt   = wv & 3;
    const int nc   = blockIdx.x;        // 0..63

    // ---- load this block's B fragments into LDS (once) ----
#pragma unroll
    for (int g = 0; g < 4; ++g) {
        const v8h* s = (const v8h*)(Wp + ((size_t)(g * 64 + nc)) * 32768);
        v8h* d = (v8h*)(Bl + (size_t)g * 32768);
        for (int i = tid; i < 4096; i += NTHR) d[i] = s[i];
    }

    const _Float16* Bwave = Bl + (size_t)gate * 32768;   // + kt*512 + lane*16

    const int sm  = tid >> 3;           // staging row 0..63
    const int sc8 = (tid & 7) * 8;      // staging col {0,8,...,56}

    const int fm   = lane & 15;                  // A-frag row within tile
    const int kh   = (lane >> 4) * 16;           // 0 or 16 (k sub-offset)
    const int arow = (mt * 16 + fm) * 72;

    // ---- per-thread persistent state: c and biases ----
    float creg[2] = {0.0f, 0.0f};
    float bgi[2], bgf[2], bgg[2], bgo[2];
#pragma unroll
    for (int e = 0; e < 2; ++e) {
        const int col = nc * 16 + ((tid * 2 + e) & 15);
        bgi[e] = bias[col];
        bgf[e] = bias[H_SZ + col];
        bgg[e] = bias[2 * H_SZ + col];
        bgo[e] = bias[3 * H_SZ + col];
    }
    __syncthreads();   // B cache ready

    int hp = 0;
    for (int t = 0; t < T_SZ; ++t) {
        const _Float16* hsrc = hbuf + (size_t)hp * (B_SZ * H_SZ);
        v8f acc = {};

        // prologue: stage K-chunk 0 (always from x)
        {
            const float* src = x + (size_t)sm * (T_SZ * I_SZ) + (size_t)t * I_SZ + sc8;
            f4 a = *(const f4*)src;
            f4 b = *(const f4*)(src + 4);
            v8h st;
            st[0] = (_Float16)a[0]; st[1] = (_Float16)a[1];
            st[2] = (_Float16)a[2]; st[3] = (_Float16)a[3];
            st[4] = (_Float16)b[0]; st[5] = (_Float16)b[1];
            st[6] = (_Float16)b[2]; st[7] = (_Float16)b[3];
            *(v8h*)&As[0][sm * 72 + sc8] = st;
        }
        __syncthreads();

        for (int kt2 = 0; kt2 < 32; ++kt2) {
            const int p = kt2 & 1;
            // prefetch next 64-wide K chunk into registers
            f4 sa, sb; v8h shh;
            const bool have = (kt2 + 1 < 32);
            bool fromx = false;
            if (have) {
                const int g0 = (kt2 + 1) * 64 + sc8;   // uniform x/h split per chunk
                fromx = (g0 < I_SZ);
                if (fromx) {
                    const float* src = x + (size_t)sm * (T_SZ * I_SZ)
                                         + (size_t)t * I_SZ + g0;
                    sa = *(const f4*)src; sb = *(const f4*)(src + 4);
                } else {
                    shh = *(const v8h*)(hsrc + (size_t)sm * H_SZ + (g0 - I_SZ));
                }
            }
            // two WMMAs per chunk, B and A both from LDS
#pragma unroll
            for (int s = 0; s < 2; ++s) {
                const int kt = kt2 * 2 + s;
                const _Float16* bp = Bwave + (size_t)kt * 512 + (size_t)lane * 16;
                v16h bf = mk16(*(const v8h*)bp, *(const v8h*)(bp + 8));
                const _Float16* ab = &As[p][arow + s * 32 + kh];
                v16h af = mk16(*(const v8h*)ab, *(const v8h*)(ab + 8));
                acc = __builtin_amdgcn_wmma_f32_16x16x32_f16(false, af, false, bf,
                                                             (short)0, acc, false, false);
            }
            if (have) {
                v8h st;
                if (fromx) {
                    st[0] = (_Float16)sa[0]; st[1] = (_Float16)sa[1];
                    st[2] = (_Float16)sa[2]; st[3] = (_Float16)sa[3];
                    st[4] = (_Float16)sb[0]; st[5] = (_Float16)sb[1];
                    st[6] = (_Float16)sb[2]; st[7] = (_Float16)sb[3];
                } else {
                    st = shh;
                }
                *(v8h*)&As[1 - p][sm * 72 + sc8] = st;
            }
            __syncthreads();
        }

        // scatter C tile (VGPR r: M = mt*16 + r + 8*(lane>=16), N = lane&15)
        {
            const int mofs = (lane >> 4) * 8;
            float* Gg = &G[gate * (64 * 17)];
#pragma unroll
            for (int r = 0; r < 8; ++r)
                Gg[(mt * 16 + mofs + r) * 17 + fm] = acc[r];
        }
        __syncthreads();

        // elementwise gates + state update (c stays in registers)
        _Float16* hdst = hbuf + (size_t)(1 - hp) * (B_SZ * H_SZ);
#pragma unroll
        for (int e = 0; e < 2; ++e) {
            const int idx = tid * 2 + e;
            const int m = idx >> 4, n = idx & 15;
            const int col = nc * 16 + n;
            const float gi = G[(0 * 64 + m) * 17 + n] + bgi[e];
            const float gf = G[(1 * 64 + m) * 17 + n] + bgf[e];
            const float gg = G[(2 * 64 + m) * 17 + n] + bgg[e];
            const float go = G[(3 * 64 + m) * 17 + n] + bgo[e];
            const float it = 1.0f / (1.0f + __expf(-gi));
            const float ft = 1.0f / (1.0f + __expf(-gf));
            const float gt = tanhf(gg);
            const float ot = 1.0f / (1.0f + __expf(-go));
            const float cn = ft * creg[e] + it * gt;
            const float hn = ot * tanhf(cn);
            creg[e] = cn;
            const size_t ci = (size_t)m * H_SZ + col;
            hdst[ci] = (_Float16)hn;
            if (t == T_SZ - 1) { out[ci] = hn; out[B_SZ * H_SZ + ci] = cn; }
        }

        // device-wide barrier (monotonic count + generation, agent scope)
        __builtin_amdgcn_fence(__ATOMIC_RELEASE, "agent");
        __syncthreads();
        if (tid == 0) {
            unsigned prev = __hip_atomic_fetch_add(bar, 1u, __ATOMIC_ACQ_REL,
                                                   __HIP_MEMORY_SCOPE_AGENT);
            const unsigned want = (unsigned)(t + 1) * (unsigned)NBLK;
            if (prev == want - 1u) {
                __hip_atomic_store(bar + 1, want, __ATOMIC_RELEASE,
                                   __HIP_MEMORY_SCOPE_AGENT);
            } else {
                while (__hip_atomic_load(bar + 1, __ATOMIC_ACQUIRE,
                                         __HIP_MEMORY_SCOPE_AGENT) < want)
                    __builtin_amdgcn_s_sleep(2);
            }
        }
        __syncthreads();
        __builtin_amdgcn_fence(__ATOMIC_ACQUIRE, "agent");
        hp ^= 1;
    }
}

extern "C" void kernel_launch(void* const* d_in, const int* in_sizes, int n_in,
                              void* d_out, int out_size, void* d_ws, size_t ws_size,
                              hipStream_t stream) {
    (void)in_sizes; (void)n_in; (void)out_size; (void)ws_size;
    const float* x    = (const float*)d_in[0];
    const float* W_ii = (const float*)d_in[1];
    const float* W_hi = (const float*)d_in[2];
    const float* W_if = (const float*)d_in[3];
    const float* W_hf = (const float*)d_in[4];
    const float* W_ig = (const float*)d_in[5];
    const float* W_hg = (const float*)d_in[6];
    const float* W_io = (const float*)d_in[7];
    const float* W_ho = (const float*)d_in[8];
    const float* b_ii = (const float*)d_in[9];
    const float* b_hi = (const float*)d_in[10];
    const float* b_if = (const float*)d_in[11];
    const float* b_hf = (const float*)d_in[12];
    const float* b_ig = (const float*)d_in[13];
    const float* b_hg = (const float*)d_in[14];
    const float* b_io = (const float*)d_in[15];
    const float* b_ho = (const float*)d_in[16];

    char* w = (char*)d_ws;
    _Float16* Wp   = (_Float16*)(w);                         // 16 MB packed weights
    float* bias    = (float*)(w + 16777216);                 // 16 KB fused bias
    _Float16* hbuf = (_Float16*)(w + 16777216 + 16384);      // 256 KB h ping-pong (f16)
    unsigned* bar  = (unsigned*)(w + 16777216 + 16384 + 262144);

    pack_w<<<2048, 256, 0, stream>>>(W_ii, W_if, W_ig, W_io,
                                     W_hi, W_hf, W_hg, W_ho, Wp);
    init_misc<<<256, 256, 0, stream>>>(b_ii, b_if, b_ig, b_io,
                                       b_hi, b_hf, b_hg, b_ho,
                                       bias, hbuf, bar);
    lstm_persistent<<<NBLK, NTHR, 0, stream>>>(x, Wp, bias, hbuf,
                                               (float*)d_out, bar);
}
